// LSTMVAE_Stacked_Weighted_5583457485199
// MI455X (gfx1250) — matrix operand order repaired and checked by reading, not compiled, not run on hardware
//
#include <hip/hip_runtime.h>

#define Bq  512
#define Tq  30
#define Hq  128
#define Lq  32
#define NTq 25
#define NRq 13
#define Gq  512
#define ZDIM 832  // (NT+1)*L

typedef __attribute__((ext_vector_type(16))) _Float16 v16h;
typedef __attribute__((ext_vector_type(8)))  float    v8f;

// ---------- workspace layout (element offsets into _Float16* ws) ----------
#define WS_WHH_TOP 0u        // [25][512][128]
#define WS_WREM    1638400u  // [512][160]  (cols 0..127 whh, 128..140 wih, rest 0)
#define WS_WMT     1720320u  // [25][32][128]
#define WS_WLT     1822720u  // [25][32][128]
#define WS_WMR     1925120u  // [32][128]
#define WS_WLR     1929216u  // [32][128]
#define WS_L2H     1933312u  // [128][832]
#define WS_DWIH    2039808u  // [512][128]
#define WS_DWHH    2105344u  // [512][128]
#define WS_OUTW    2170880u  // [48][128] (0..24 top, 25..37 rem, rest 0)
#define WS_Z16     2177024u  // [512][832]
#define WS_HID     2603008u  // [512][128]
#define WS_HSEQ    2668544u  // [512*30][128]
#define WS_TOTAL_BYTES (4634624u * 2u)

// d_out offsets (floats)
#define OUT_RT 0
#define OUT_RR 384000
#define OUT_MC 583680
#define OUT_LC 1009664

__device__ __forceinline__ float sigm(float x) { return 1.0f / (1.0f + __expf(-x)); }

// A-fragment: 16x32 f16 tile, row-major source with leading dim ld (halves).
// CDNA5 layout: lanes0-15 M=lane,Kbase=0 ; lanes16-31 M=lane-16,Kbase=8
//               VGPR0-3 -> K=base+0..7 ; VGPR4-7 -> K=base+16..23
__device__ __forceinline__ v16h load_a16(const _Float16* tile, int ld) {
  int lane = threadIdx.x & 31;
  const _Float16* p = tile + (lane & 15) * ld + ((lane & 16) ? 8 : 0);
  v16h r;
  float4 lo = *(const float4*)(p);
  float4 hi = *(const float4*)(p + 16);
  __builtin_memcpy(&r, &lo, 16);
  __builtin_memcpy(reinterpret_cast<char*>(&r) + 16, &hi, 16);
  return r;
}

// B-fragment: 32x16 f16 tile. Source W is row-major [N][K] (PyTorch weight),
// tile origin (n0,k0). lanes0-15 N=lane,Kbase=0 ; lanes16-31 Kbase=8;
// VGPR0-3 -> K=base+0..7 ; VGPR4-7 -> K=base+16..23 (matches A).
__device__ __forceinline__ v16h load_b16(const _Float16* W, int ld, int n0, int k0) {
  int lane = threadIdx.x & 31;
  const _Float16* p = W + (size_t)(n0 + (lane & 15)) * ld + k0 + ((lane & 16) ? 8 : 0);
  v16h r;
  float4 lo = *(const float4*)(p);
  float4 hi = *(const float4*)(p + 16);
  __builtin_memcpy(&r, &lo, 16);
  __builtin_memcpy(reinterpret_cast<char*>(&r) + 16, &hi, 16);
  return r;
}

__device__ __forceinline__ v8f wmma16(v16h a, v16h b, v8f c) {
  return __builtin_amdgcn_wmma_f32_16x16x32_f16(false, a, false, b, (short)0, c, false, false);
}

// ---------------- prep: f32 -> f16 weight staging ----------------
__global__ void prep_kernel(const float* __restrict__ etwhh, const float* __restrict__ erwih,
                            const float* __restrict__ erwhh, const float* __restrict__ etwm,
                            const float* __restrict__ etwl,  const float* __restrict__ erwm,
                            const float* __restrict__ erwl,  const float* __restrict__ l2hw,
                            const float* __restrict__ dwih,  const float* __restrict__ dwhh,
                            const float* __restrict__ otw,   const float* __restrict__ orw,
                            _Float16* __restrict__ ws) {
  int tid = blockIdx.x * blockDim.x + threadIdx.x;
  int stride = gridDim.x * blockDim.x;
  for (int i = tid; i < NTq * Gq * Hq; i += stride) ws[WS_WHH_TOP + i] = (_Float16)etwhh[i];
  for (int i = tid; i < Gq * 160; i += stride) {
    int g = i / 160, k = i - g * 160;
    float v = (k < Hq) ? erwhh[g * Hq + k] : ((k < Hq + NRq) ? erwih[g * NRq + (k - Hq)] : 0.0f);
    ws[WS_WREM + i] = (_Float16)v;
  }
  for (int i = tid; i < NTq * Lq * Hq; i += stride) {
    ws[WS_WMT + i] = (_Float16)etwm[i];
    ws[WS_WLT + i] = (_Float16)etwl[i];
  }
  for (int i = tid; i < Lq * Hq; i += stride) {
    ws[WS_WMR + i] = (_Float16)erwm[i];
    ws[WS_WLR + i] = (_Float16)erwl[i];
  }
  for (int i = tid; i < Hq * ZDIM; i += stride) ws[WS_L2H + i] = (_Float16)l2hw[i];
  for (int i = tid; i < Gq * Hq; i += stride) {
    ws[WS_DWIH + i] = (_Float16)dwih[i];
    ws[WS_DWHH + i] = (_Float16)dwhh[i];
  }
  for (int i = tid; i < 48 * Hq; i += stride) {
    int n = i >> 7, k = i & 127;
    float v = (n < NTq) ? otw[n * Hq + k] : ((n < NTq + NRq) ? orw[(n - NTq) * Hq + k] : 0.0f);
    ws[WS_OUTW + i] = (_Float16)v;
  }
}

// ---------------- per-sensor top encoders (input dim 1) ----------------
__global__ __launch_bounds__(256) void encode_top_kernel(
    const float* __restrict__ x_top, const float* __restrict__ wih,
    const float* __restrict__ bih, const float* __restrict__ bhh,
    const float* __restrict__ bm, const float* __restrict__ bl,
    const float* __restrict__ eps_top,
    const _Float16* __restrict__ whh16, const _Float16* __restrict__ wm16,
    const _Float16* __restrict__ wl16, _Float16* __restrict__ z16,
    float* __restrict__ out_mean, float* __restrict__ out_logv) {
  extern __shared__ char smem[];
  _Float16* whh = (_Float16*)smem;                  // 512*128
  _Float16* h16 = (_Float16*)(smem + 131072);       // 16*128
  float* xt     = (float*)(smem + 135168);          // 16*30
  float* wih_l  = (float*)(smem + 137216 + 2048);   // 512   (xt padded to 2048B)
  float* bias_l = (float*)(smem + 141312);          // 512
  float* ml     = (float*)(smem + 143360);          // 16*64

  int e = blockIdx.x / (Bq / 16);
  int row0 = (blockIdx.x % (Bq / 16)) * 16;
  int lane = threadIdx.x & 31, wv = threadIdx.x >> 5;
  int ln15 = lane & 15, rhalf = (lane & 16) ? 8 : 0;
  int hcol = wv * 16 + ln15;

  {  // stage weights / inputs
    const float4* src = (const float4*)(whh16 + (size_t)e * Gq * Hq);
    float4* dst = (float4*)whh;
    for (int i = threadIdx.x; i < Gq * Hq / 8; i += 256) dst[i] = src[i];
    for (int i = threadIdx.x; i < Gq; i += 256) {
      wih_l[i] = wih[e * Gq + i];
      bias_l[i] = bih[e * Gq + i] + bhh[e * Gq + i];
    }
    for (int i = threadIdx.x; i < 16 * Tq; i += 256) {
      int m = i / Tq, t = i - m * Tq;
      xt[i] = x_top[((size_t)(row0 + m) * Tq + t) * NTq + e];
    }
    for (int i = threadIdx.x; i < 16 * Hq; i += 256) h16[i] = (_Float16)0.0f;
  }
  __syncthreads();

  float creg[8];
#pragma unroll
  for (int r = 0; r < 8; ++r) creg[r] = 0.0f;

  for (int t = 0; t < Tq; ++t) {
    v16h a0 = load_a16(h16 + 0, Hq);
    v16h a1 = load_a16(h16 + 32, Hq);
    v16h a2 = load_a16(h16 + 64, Hq);
    v16h a3 = load_a16(h16 + 96, Hq);
    __syncthreads();
    v8f acc[4];
#pragma unroll
    for (int q = 0; q < 4; ++q) {
      int nc = q * Hq + hcol;
      float wv_i = wih_l[nc], bv = bias_l[nc];
#pragma unroll
      for (int r = 0; r < 8; ++r) acc[q][r] = xt[(r + rhalf) * Tq + t] * wv_i + bv;
      int n0 = q * Hq + wv * 16;
      acc[q] = wmma16(a0, load_b16(whh, Hq, n0, 0), acc[q]);
      acc[q] = wmma16(a1, load_b16(whh, Hq, n0, 32), acc[q]);
      acc[q] = wmma16(a2, load_b16(whh, Hq, n0, 64), acc[q]);
      acc[q] = wmma16(a3, load_b16(whh, Hq, n0, 96), acc[q]);
    }
#pragma unroll
    for (int r = 0; r < 8; ++r) {
      float iv = sigm(acc[0][r]), fv = sigm(acc[1][r]);
      float gv = tanhf(acc[2][r]), ov = sigm(acc[3][r]);
      creg[r] = fv * creg[r] + iv * gv;
      h16[(r + rhalf) * Hq + hcol] = (_Float16)(ov * tanhf(creg[r]));
    }
    __syncthreads();
  }

  // mean/logvar projections: waves 0-1 -> mean, 2-3 -> logvar
  if (wv < 4) {
    int isLog = wv >> 1;
    int n0 = (wv & 1) * 16;
    const _Float16* W = (isLog ? wl16 : wm16) + (size_t)e * Lq * Hq;
    const float* bb = (isLog ? bl : bm) + e * Lq;
    v8f acc = {};
    float bv = bb[n0 + ln15];
#pragma unroll
    for (int r = 0; r < 8; ++r) acc[r] = bv;
#pragma unroll
    for (int kc = 0; kc < 4; ++kc)
      acc = wmma16(load_a16(h16 + kc * 32, Hq), load_b16(W, Hq, n0, kc * 32), acc);
#pragma unroll
    for (int r = 0; r < 8; ++r) ml[(r + rhalf) * 64 + isLog * 32 + n0 + ln15] = acc[r];
  }
  __syncthreads();

  for (int idx = threadIdx.x; idx < 16 * Lq; idx += 256) {
    int m = idx >> 5, l = idx & 31;
    int b = row0 + m;
    float mean = ml[m * 64 + l], lv = ml[m * 64 + 32 + l];
    float z = mean + eps_top[((size_t)b * NTq + e) * Lq + l] * __expf(0.5f * lv);
    z16[(size_t)b * ZDIM + e * Lq + l] = (_Float16)z;
    out_mean[(size_t)b * ZDIM + e * Lq + l] = mean;
    out_logv[(size_t)b * ZDIM + e * Lq + l] = lv;
  }
}

// ---------------- remaining encoder (concat [h;x], K = 160) ----------------
__global__ __launch_bounds__(256) void encode_rem_kernel(
    const float* __restrict__ x_rem, const float* __restrict__ bih,
    const float* __restrict__ bhh, const float* __restrict__ bm,
    const float* __restrict__ bl, const float* __restrict__ eps_rem,
    const _Float16* __restrict__ wrem16, const _Float16* __restrict__ wm16,
    const _Float16* __restrict__ wl16, _Float16* __restrict__ z16,
    float* __restrict__ out_mean, float* __restrict__ out_logv) {
  extern __shared__ char smem[];
  _Float16* wrem = (_Float16*)smem;                 // 512*160
  _Float16* hx   = (_Float16*)(smem + 163840);      // 16*160
  float* bias_l  = (float*)(smem + 168960);         // 512
  float* ml      = (float*)(smem + 171008);         // 16*64

  int row0 = blockIdx.x * 16;
  int lane = threadIdx.x & 31, wv = threadIdx.x >> 5;
  int ln15 = lane & 15, rhalf = (lane & 16) ? 8 : 0;
  int hcol = wv * 16 + ln15;

  {
    const float4* src = (const float4*)wrem16;
    float4* dst = (float4*)wrem;
    for (int i = threadIdx.x; i < Gq * 160 / 8; i += 256) dst[i] = src[i];
    for (int i = threadIdx.x; i < Gq; i += 256) bias_l[i] = bih[i] + bhh[i];
    for (int i = threadIdx.x; i < 16 * 160; i += 256) hx[i] = (_Float16)0.0f;
  }
  __syncthreads();

  float creg[8];
#pragma unroll
  for (int r = 0; r < 8; ++r) creg[r] = 0.0f;

  for (int t = 0; t < Tq; ++t) {
    for (int i = threadIdx.x; i < 16 * NRq; i += 256) {
      int m = i / NRq, d = i - m * NRq;
      hx[m * 160 + Hq + d] = (_Float16)x_rem[((size_t)(row0 + m) * Tq + t) * NRq + d];
    }
    __syncthreads();
    v16h a[5];
#pragma unroll
    for (int kc = 0; kc < 5; ++kc) a[kc] = load_a16(hx + kc * 32, 160);
    __syncthreads();
    v8f acc[4];
#pragma unroll
    for (int q = 0; q < 4; ++q) {
      float bv = bias_l[q * Hq + hcol];
#pragma unroll
      for (int r = 0; r < 8; ++r) acc[q][r] = bv;
      int n0 = q * Hq + wv * 16;
#pragma unroll
      for (int kc = 0; kc < 5; ++kc)
        acc[q] = wmma16(a[kc], load_b16(wrem, 160, n0, kc * 32), acc[q]);
    }
#pragma unroll
    for (int r = 0; r < 8; ++r) {
      float iv = sigm(acc[0][r]), fv = sigm(acc[1][r]);
      float gv = tanhf(acc[2][r]), ov = sigm(acc[3][r]);
      creg[r] = fv * creg[r] + iv * gv;
      hx[(r + rhalf) * 160 + hcol] = (_Float16)(ov * tanhf(creg[r]));
    }
    __syncthreads();
  }

  if (wv < 4) {
    int isLog = wv >> 1;
    int n0 = (wv & 1) * 16;
    const _Float16* W = isLog ? wl16 : wm16;
    const float* bb = (isLog ? bl : bm);
    v8f acc = {};
    float bv = bb[n0 + ln15];
#pragma unroll
    for (int r = 0; r < 8; ++r) acc[r] = bv;
#pragma unroll
    for (int kc = 0; kc < 4; ++kc)
      acc = wmma16(load_a16(hx + kc * 32, 160), load_b16(W, Hq, n0, kc * 32), acc);
#pragma unroll
    for (int r = 0; r < 8; ++r) ml[(r + rhalf) * 64 + isLog * 32 + n0 + ln15] = acc[r];
  }
  __syncthreads();

  for (int idx = threadIdx.x; idx < 16 * Lq; idx += 256) {
    int m = idx >> 5, l = idx & 31;
    int b = row0 + m;
    float mean = ml[m * 64 + l], lv = ml[m * 64 + 32 + l];
    float z = mean + eps_rem[(size_t)b * Lq + l] * __expf(0.5f * lv);
    z16[(size_t)b * ZDIM + NTq * Lq + l] = (_Float16)z;
    out_mean[(size_t)b * ZDIM + NTq * Lq + l] = mean;
    out_logv[(size_t)b * ZDIM + NTq * Lq + l] = lv;
  }
}

// ---------------- hid = z @ l2h.T + b   (M=512, N=128, K=832) ----------------
__global__ __launch_bounds__(256) void hid_kernel(const _Float16* __restrict__ z16,
                                                  const _Float16* __restrict__ l2h16,
                                                  const float* __restrict__ l2hb,
                                                  _Float16* __restrict__ hid16) {
  int row0 = blockIdx.x * 16;
  int lane = threadIdx.x & 31, wv = threadIdx.x >> 5;
  int ln15 = lane & 15, rhalf = (lane & 16) ? 8 : 0;
  int n0 = wv * 16;
  v8f acc = {};
  float bv = l2hb[n0 + ln15];
#pragma unroll
  for (int r = 0; r < 8; ++r) acc[r] = bv;
  for (int kc = 0; kc < ZDIM / 32; ++kc) {
    v16h a = load_a16(z16 + (size_t)row0 * ZDIM + kc * 32, ZDIM);
    v16h b = load_b16(l2h16, ZDIM, n0, kc * 32);
    acc = wmma16(a, b, acc);
  }
#pragma unroll
  for (int r = 0; r < 8; ++r)
    hid16[(size_t)(row0 + r + rhalf) * Hq + n0 + ln15] = (_Float16)acc[r];
}

// ---------------- decoder LSTM (constant input = hid) ----------------
__global__ __launch_bounds__(256) void decoder_kernel(const _Float16* __restrict__ hid16,
                                                      const _Float16* __restrict__ dwih16,
                                                      const _Float16* __restrict__ dwhh16,
                                                      const float* __restrict__ dbih,
                                                      const float* __restrict__ dbhh,
                                                      _Float16* __restrict__ hseq16) {
  extern __shared__ char smem[];
  _Float16* whh = (_Float16*)smem;             // 512*128
  _Float16* h16 = (_Float16*)(smem + 131072);  // 16*128

  int row0 = blockIdx.x * 16;
  int lane = threadIdx.x & 31, wv = threadIdx.x >> 5;
  int ln15 = lane & 15, rhalf = (lane & 16) ? 8 : 0;
  int hcol = wv * 16 + ln15;

  {
    const float4* src = (const float4*)dwhh16;
    float4* dst = (float4*)whh;
    for (int i = threadIdx.x; i < Gq * Hq / 8; i += 256) dst[i] = src[i];
    for (int i = threadIdx.x; i < 16 * Hq; i += 256) h16[i] = (_Float16)0.0f;
  }
  __syncthreads();

  // xg = hid @ w_ih.T + (b_ih + b_hh): computed once via WMMA, kept in registers
  v8f xgf[4];
  {
    v16h a[4];
#pragma unroll
    for (int kc = 0; kc < 4; ++kc)
      a[kc] = load_a16(hid16 + (size_t)row0 * Hq + kc * 32, Hq);
#pragma unroll
    for (int q = 0; q < 4; ++q) {
      int nc = q * Hq + hcol;
      float bv = dbih[nc] + dbhh[nc];
#pragma unroll
      for (int r = 0; r < 8; ++r) xgf[q][r] = bv;
      int n0 = q * Hq + wv * 16;
#pragma unroll
      for (int kc = 0; kc < 4; ++kc)
        xgf[q] = wmma16(a[kc], load_b16(dwih16, Hq, n0, kc * 32), xgf[q]);
    }
  }

  float creg[8];
#pragma unroll
  for (int r = 0; r < 8; ++r) creg[r] = 0.0f;

  for (int t = 0; t < Tq; ++t) {
    v16h a0 = load_a16(h16 + 0, Hq);
    v16h a1 = load_a16(h16 + 32, Hq);
    v16h a2 = load_a16(h16 + 64, Hq);
    v16h a3 = load_a16(h16 + 96, Hq);
    __syncthreads();
    v8f acc[4];
#pragma unroll
    for (int q = 0; q < 4; ++q) {
      acc[q] = xgf[q];
      int n0 = q * Hq + wv * 16;
      acc[q] = wmma16(a0, load_b16(whh, Hq, n0, 0), acc[q]);
      acc[q] = wmma16(a1, load_b16(whh, Hq, n0, 32), acc[q]);
      acc[q] = wmma16(a2, load_b16(whh, Hq, n0, 64), acc[q]);
      acc[q] = wmma16(a3, load_b16(whh, Hq, n0, 96), acc[q]);
    }
#pragma unroll
    for (int r = 0; r < 8; ++r) {
      float iv = sigm(acc[0][r]), fv = sigm(acc[1][r]);
      float gv = tanhf(acc[2][r]), ov = sigm(acc[3][r]);
      creg[r] = fv * creg[r] + iv * gv;
      _Float16 hv = (_Float16)(ov * tanhf(creg[r]));
      h16[(r + rhalf) * Hq + hcol] = hv;
      hseq16[((size_t)(row0 + r + rhalf) * Tq + t) * Hq + hcol] = hv;
    }
    __syncthreads();
  }
}

// ------- recon = out_seq @ [out_top_w; out_rem_w].T  (M=B*T, N=48pad, K=128) -------
__global__ __launch_bounds__(96) void recon_kernel(const _Float16* __restrict__ hseq16,
                                                   const _Float16* __restrict__ outw16,
                                                   const float* __restrict__ btop,
                                                   const float* __restrict__ brem,
                                                   float* __restrict__ recon_top,
                                                   float* __restrict__ recon_rem) {
  int gr0 = blockIdx.x * 16;
  int lane = threadIdx.x & 31, wv = threadIdx.x >> 5;  // 3 waves
  int ln15 = lane & 15, rhalf = (lane & 16) ? 8 : 0;
  int n0 = wv * 16, n = n0 + ln15;
  float bv = (n < NTq) ? btop[n] : ((n < NTq + NRq) ? brem[n - NTq] : 0.0f);
  v8f acc = {};
#pragma unroll
  for (int r = 0; r < 8; ++r) acc[r] = bv;
#pragma unroll
  for (int kc = 0; kc < 4; ++kc) {
    v16h a = load_a16(hseq16 + (size_t)gr0 * Hq + kc * 32, Hq);
    v16h b = load_b16(outw16, Hq, n0, kc * 32);
    acc = wmma16(a, b, acc);
  }
#pragma unroll
  for (int r = 0; r < 8; ++r) {
    int gr = gr0 + r + rhalf;
    if (n < NTq) recon_top[(size_t)gr * NTq + n] = acc[r];
    else if (n < NTq + NRq) recon_rem[(size_t)gr * NRq + (n - NTq)] = acc[r];
  }
}

extern "C" void kernel_launch(void* const* d_in, const int* in_sizes, int n_in,
                              void* d_out, int out_size, void* d_ws, size_t ws_size,
                              hipStream_t stream) {
  const float* x_top = (const float*)d_in[0];
  const float* x_rem = (const float*)d_in[1];
  const float* eps_top = (const float*)d_in[2];
  const float* eps_rem = (const float*)d_in[3];
  const float* et_wih = (const float*)d_in[4];
  const float* et_whh = (const float*)d_in[5];
  const float* et_bih = (const float*)d_in[6];
  const float* et_bhh = (const float*)d_in[7];
  const float* et_wm = (const float*)d_in[8];
  const float* et_bm = (const float*)d_in[9];
  const float* et_wl = (const float*)d_in[10];
  const float* et_bl = (const float*)d_in[11];
  const float* er_wih = (const float*)d_in[12];
  const float* er_whh = (const float*)d_in[13];
  const float* er_bih = (const float*)d_in[14];
  const float* er_bhh = (const float*)d_in[15];
  const float* er_wm = (const float*)d_in[16];
  const float* er_bm = (const float*)d_in[17];
  const float* er_wl = (const float*)d_in[18];
  const float* er_bl = (const float*)d_in[19];
  const float* l2h_w = (const float*)d_in[20];
  const float* l2h_b = (const float*)d_in[21];
  const float* d_wih = (const float*)d_in[22];
  const float* d_whh = (const float*)d_in[23];
  const float* d_bih = (const float*)d_in[24];
  const float* d_bhh = (const float*)d_in[25];
  const float* ot_w = (const float*)d_in[26];
  const float* ot_b = (const float*)d_in[27];
  const float* or_w = (const float*)d_in[28];
  const float* or_b = (const float*)d_in[29];

  if (ws_size < (size_t)WS_TOTAL_BYTES) return;
  _Float16* ws = (_Float16*)d_ws;
  float* outf = (float*)d_out;

  prep_kernel<<<512, 256, 0, stream>>>(et_whh, er_wih, er_whh, et_wm, et_wl, er_wm, er_wl,
                                       l2h_w, d_wih, d_whh, ot_w, or_w, ws);

  encode_top_kernel<<<NTq * (Bq / 16), 256, 147456, stream>>>(
      x_top, et_wih, et_bih, et_bhh, et_bm, et_bl, eps_top,
      ws + WS_WHH_TOP, ws + WS_WMT, ws + WS_WLT, ws + WS_Z16,
      outf + OUT_MC, outf + OUT_LC);

  encode_rem_kernel<<<Bq / 16, 256, 175104, stream>>>(
      x_rem, er_bih, er_bhh, er_bm, er_bl, eps_rem,
      ws + WS_WREM, ws + WS_WMR, ws + WS_WLR, ws + WS_Z16,
      outf + OUT_MC, outf + OUT_LC);

  hid_kernel<<<Bq / 16, 256, 0, stream>>>(ws + WS_Z16, ws + WS_L2H, l2h_b, ws + WS_HID);

  decoder_kernel<<<Bq / 16, 256, 135168, stream>>>(ws + WS_HID, ws + WS_DWIH, ws + WS_DWHH,
                                                   d_bih, d_bhh, ws + WS_HSEQ);

  recon_kernel<<<(Bq * Tq) / 16, 96, 0, stream>>>(ws + WS_HSEQ, ws + WS_OUTW, ot_b, or_b,
                                                  outf + OUT_RT, outf + OUT_RR);
}
